// GraphBiasedAttention_63436666962367
// MI455X (gfx1250) — matrix operand order repaired and checked by reading, not compiled, not run on hardware
//
#include <hip/hip_runtime.h>
#include <hip/hip_bf16.h>

// ---------------------------------------------------------------------------
// GraphBiasedAttention for MI455X (gfx1250): bf16 WMMA pipeline
//   B=16, C=1024, D=512, H=8, HD=64
// ---------------------------------------------------------------------------

#define B_  16
#define C_  1024
#define D_  512
#define H_  8
#define HD_ 64

typedef __attribute__((ext_vector_type(16))) __bf16 v16bf;
typedef __attribute__((ext_vector_type(8)))  __bf16 v8bf;
typedef __attribute__((ext_vector_type(4)))  __bf16 v4bf;
typedef __attribute__((ext_vector_type(8)))  float  v8f;

union AB16 { v16bf v; v8bf h[2]; };

__device__ __forceinline__ __bf16 f2bf(float f) {
    unsigned u = __builtin_bit_cast(unsigned, f);
    u += 0x7FFFu + ((u >> 16) & 1u);              // round-to-nearest-even
    unsigned short hv = (unsigned short)(u >> 16);
    return __builtin_bit_cast(__bf16, hv);
}

__device__ __forceinline__ v8f wmma_bf16(const v16bf& a, const v16bf& b, v8f c) {
    // emits v_wmma_f32_16x16x32_bf16
    return __builtin_amdgcn_wmma_f32_16x16x32_bf16(
        false, a, false, b, (short)0, c, false, false);
}

__device__ __forceinline__ v8f v8f_zero() {
    v8f z = {0.f, 0.f, 0.f, 0.f, 0.f, 0.f, 0.f, 0.f};
    return z;
}

// ---------------------------------------------------------------------------
// Stage 1a: fp32 -> bf16 elementwise convert
// ---------------------------------------------------------------------------
__global__ void k_convert_bf16(const float* __restrict__ src,
                               __bf16* __restrict__ dst, int n) {
    int i = blockIdx.x * blockDim.x + threadIdx.x;
    if (i < n) dst[i] = f2bf(src[i]);
}

// Stage 1b: transpose [K][N] fp32 -> [N][K] bf16 (so B fragments load contig)
__global__ void k_transpose_bf16(const float* __restrict__ src,
                                 __bf16* __restrict__ dst, int K, int N) {
    int i = blockIdx.x * blockDim.x + threadIdx.x;   // i = n*K + k
    if (i < K * N) {
        int n = i / K, k = i - n * K;
        dst[i] = f2bf(src[k * N + n]);
    }
}

// ---------------------------------------------------------------------------
// Stage 2: QKV GEMM  [16384 x 512] @ [512 x 1536] + bias
//   block = 128 threads (4 waves); block tile 64(M) x 64(N); wave: 16x64
//   q,k -> [B,H,C,HD] bf16 ; v -> [B,H,HD,C] bf16 (transposed)
// ---------------------------------------------------------------------------
__global__ __launch_bounds__(128) void k_qkv_gemm(
    const __bf16* __restrict__ xbf,      // [16384][512]
    const __bf16* __restrict__ wT,       // [1536][512]
    const float*  __restrict__ bias,     // [1536]
    __bf16* __restrict__ qs, __bf16* __restrict__ ks, __bf16* __restrict__ vT)
{
    const int lane = threadIdx.x & 31;
    const int wave = threadIdx.x >> 5;
    const int m0   = blockIdx.y * 64 + wave * 16;
    const int n0   = blockIdx.x * 64;
    const int mRow = m0 + (lane & 15);
    const int aoff = (lane < 16) ? 0 : 8;
    const int boff = (lane < 16) ? 0 : 8;
    const int rbase = (lane >> 4) << 3;

    v8f acc[4];
#pragma unroll
    for (int nt = 0; nt < 4; ++nt) acc[nt] = v8f_zero();

    for (int kk = 0; kk < 512; kk += 32) {
        AB16 a;
        const __bf16* ap = xbf + (size_t)mRow * 512 + kk + aoff;
        a.h[0] = *(const v8bf*)(ap);
        a.h[1] = *(const v8bf*)(ap + 16);
#pragma unroll
        for (int nt = 0; nt < 4; ++nt) {
            const int n = n0 + nt * 16 + (lane & 15);
            AB16 b;
            const __bf16* bp = wT + (size_t)n * 512 + kk + boff;
            b.h[0] = *(const v8bf*)(bp);
            b.h[1] = *(const v8bf*)(bp + 16);
            acc[nt] = wmma_bf16(a.v, b.v, acc[nt]);
        }
    }

    const int which = n0 >> 9;            // 0=q 1=k 2=v (uniform per block)
    const int h     = (n0 >> 6) & 7;      // uniform per block
#pragma unroll
    for (int nt = 0; nt < 4; ++nt) {
        const int n  = n0 + nt * 16 + (lane & 15);
        const int hd = n & 63;
        const float bv = bias[n];
#pragma unroll
        for (int r = 0; r < 8; ++r) {
            const int mg = m0 + rbase + r;
            const int bb = mg >> 10;            // batch
            const int cc = mg & (C_ - 1);       // channel
            const __bf16 o = f2bf(acc[nt][r] + bv);
            if (which == 0)
                qs[(((size_t)bb * H_ + h) * C_ + cc) * HD_ + hd] = o;
            else if (which == 1)
                ks[(((size_t)bb * H_ + h) * C_ + cc) * HD_ + hd] = o;
            else
                vT[(((size_t)bb * H_ + h) * HD_ + hd) * C_ + cc] = o;
        }
    }
}

// ---------------------------------------------------------------------------
// Stage 3: fused attention per (b*H+h, 16-query tile)
//   256 threads = 8 waves. Waves 0..7 each own a 128-key strip for QK^T.
//   Softmax in LDS (vectorized float4); fp32 attn streamed to d_out with
//   b128 stores; bf16 probs in LDS. PV matmul split-K across all 8 waves.
// ---------------------------------------------------------------------------
__global__ __launch_bounds__(256) void k_attn(
    const __bf16* __restrict__ qs, const __bf16* __restrict__ ks,
    const __bf16* __restrict__ vT, const float* __restrict__ adj,
    const float*  __restrict__ lambda_p,
    float* __restrict__ attn_out,        // [B,H,C,C] fp32
    __bf16* __restrict__ ctx)            // [B,C,D] bf16
{
    __shared__ __align__(16) float  s_scores[16][C_];    // 64 KB (reused)
    __shared__ __align__(16) __bf16 s_probs[16][C_];     // 32 KB
    __shared__ float s_red[16][16];
    __shared__ float s_rowmax[16];
    __shared__ float s_rowsum[16];

    const int lane = threadIdx.x & 31;
    const int wave = threadIdx.x >> 5;
    const int bh   = blockIdx.y;             // b*H + h
    const int q0   = blockIdx.x * 16;        // query tile base
    const float lam   = lambda_p[0];
    const float scale = 0.125f;              // 1/sqrt(64)

    const __bf16* qbase = qs + (size_t)bh * C_ * HD_;
    const __bf16* kbase = ks + (size_t)bh * C_ * HD_;
    const __bf16* vbase = vT + (size_t)bh * HD_ * C_;

    const int aoff  = (lane < 16) ? 0 : 8;
    const int boff  = (lane < 16) ? 0 : 8;
    const int rbase = (lane >> 4) << 3;

    // ---- phase 1: S = q @ k^T for 16 queries x 128 keys per wave ----------
    AB16 qf[2];
    {
        const __bf16* qp = qbase + (size_t)(q0 + (lane & 15)) * HD_;
        qf[0].h[0] = *(const v8bf*)(qp + aoff);
        qf[0].h[1] = *(const v8bf*)(qp + aoff + 16);
        qf[1].h[0] = *(const v8bf*)(qp + 32 + aoff);
        qf[1].h[1] = *(const v8bf*)(qp + 32 + aoff + 16);
    }
    v8f acc[8];
#pragma unroll
    for (int kt = 0; kt < 8; ++kt) acc[kt] = v8f_zero();

    const int key0 = wave * 128;
#pragma unroll
    for (int kt = 0; kt < 8; ++kt) {
        const int key = key0 + kt * 16 + (lane & 15);
        const __bf16* kp = kbase + (size_t)key * HD_;
#pragma unroll
        for (int kk = 0; kk < 2; ++kk) {
            AB16 b;
            b.h[0] = *(const v8bf*)(kp + kk * 32 + boff);
            b.h[1] = *(const v8bf*)(kp + kk * 32 + boff + 16);
            acc[kt] = wmma_bf16(qf[kk].v, b.v, acc[kt]);
        }
    }
    // raw scores -> LDS (scale + graph bias applied in vectorized phase 2)
#pragma unroll
    for (int kt = 0; kt < 8; ++kt) {
        const int col = key0 + kt * 16 + (lane & 15);
#pragma unroll
        for (int r = 0; r < 8; ++r)
            s_scores[rbase + r][col] = acc[kt][r];
    }
    __syncthreads();

    // ---- phase 2: bias + softmax; 16 threads/row, contiguous 64-col chunks
    {
        const int row = threadIdx.x >> 4;
        const int i   = threadIdx.x & 15;
        const int c0  = i * 64;
        const float* adjrow = adj + (size_t)(q0 + row) * C_;

        float mx = -1e30f;
        for (int c = c0; c < c0 + 64; c += 4) {
            float4 s = *(float4*)(&s_scores[row][c]);
            const float4 a = *(const float4*)(adjrow + c);   // b128 load
            s.x = s.x * scale + lam * a.x;
            s.y = s.y * scale + lam * a.y;
            s.z = s.z * scale + lam * a.z;
            s.w = s.w * scale + lam * a.w;
            *(float4*)(&s_scores[row][c]) = s;
            mx = fmaxf(mx, fmaxf(fmaxf(s.x, s.y), fmaxf(s.z, s.w)));
        }
        s_red[row][i] = mx;
        __syncthreads();
        if (i == 0) {
            float m = s_red[row][0];
#pragma unroll
            for (int j = 1; j < 16; ++j) m = fmaxf(m, s_red[row][j]);
            s_rowmax[row] = m;
        }
        __syncthreads();
        const float m = s_rowmax[row];
        float sum = 0.f;
        for (int c = c0; c < c0 + 64; c += 4) {
            float4 s = *(float4*)(&s_scores[row][c]);
            s.x = __expf(s.x - m);
            s.y = __expf(s.y - m);
            s.z = __expf(s.z - m);
            s.w = __expf(s.w - m);
            *(float4*)(&s_scores[row][c]) = s;
            sum += (s.x + s.y) + (s.z + s.w);
        }
        s_red[row][i] = sum;
        __syncthreads();
        if (i == 0) {
            float s = 0.f;
#pragma unroll
            for (int j = 0; j < 16; ++j) s += s_red[row][j];
            s_rowsum[row] = s;
        }
        __syncthreads();
        const float inv = 1.0f / s_rowsum[row];
        float* arow = attn_out + ((size_t)bh * C_ + (q0 + row)) * C_;
        for (int c = c0; c < c0 + 64; c += 4) {
            float4 s = *(float4*)(&s_scores[row][c]);
            s.x *= inv; s.y *= inv; s.z *= inv; s.w *= inv;
            *(float4*)(arow + c) = s;                        // b128 store
            v4bf p = {f2bf(s.x), f2bf(s.y), f2bf(s.z), f2bf(s.w)};
            *(v4bf*)(&s_probs[row][c]) = p;                  // ds b64 store
        }
    }
    __syncthreads();

    // ---- phase 3: context = probs @ v  (16 x 64, split-K over 8 waves) ----
    {
        v8f cacc = v8f_zero();
        const int nt = wave & 3;                  // hd tile 0..3
        const int n  = nt * 16 + (lane & 15);     // hd index
        const __bf16* vp = vbase + (size_t)n * C_;
        const int m = lane & 15;
        const int kbeg = (wave < 4) ? 0 : 512;    // split-K halves
        for (int k0 = kbeg; k0 < kbeg + 512; k0 += 32) {
            AB16 a, b;
            const __bf16* pp = &s_probs[m][k0 + aoff];
            a.h[0] = *(const v8bf*)(pp);
            a.h[1] = *(const v8bf*)(pp + 16);
            const __bf16* bp = vp + k0 + boff;
            b.h[0] = *(const v8bf*)(bp);
            b.h[1] = *(const v8bf*)(bp + 16);
            cacc = wmma_bf16(a.v, b.v, cacc);
        }
        float* spart = &s_scores[0][0];           // reuse dead LDS as scratch
        if (wave >= 4) {
#pragma unroll
            for (int r = 0; r < 8; ++r)
                spart[(nt * 16 + rbase + r) * 16 + (lane & 15)] = cacc[r];
        }
        __syncthreads();
        if (wave < 4) {
            const int bb = bh >> 3, h = bh & 7;
#pragma unroll
            for (int r = 0; r < 8; ++r) {
                const float v = cacc[r]
                    + spart[(nt * 16 + rbase + r) * 16 + (lane & 15)];
                const int cg = q0 + rbase + r;
                ctx[((size_t)(bb * C_ + cg)) * D_ + h * HD_ + n] = f2bf(v);
            }
        }
    }
}

// ---------------------------------------------------------------------------
// Stage 4: projection GEMM  [16384 x 512] @ [512 x 512] + bias -> fp32 out
// ---------------------------------------------------------------------------
__global__ __launch_bounds__(128) void k_proj_gemm(
    const __bf16* __restrict__ ctx,      // [16384][512]
    const __bf16* __restrict__ wT,       // [512][512]
    const float*  __restrict__ bias,     // [512]
    float* __restrict__ out)             // [16384][512]
{
    const int lane = threadIdx.x & 31;
    const int wave = threadIdx.x >> 5;
    const int m0   = blockIdx.y * 64 + wave * 16;
    const int n0   = blockIdx.x * 64;
    const int mRow = m0 + (lane & 15);
    const int aoff = (lane < 16) ? 0 : 8;
    const int boff = (lane < 16) ? 0 : 8;
    const int rbase = (lane >> 4) << 3;

    v8f acc[4];
#pragma unroll
    for (int nt = 0; nt < 4; ++nt) acc[nt] = v8f_zero();

    for (int kk = 0; kk < 512; kk += 32) {
        AB16 a;
        const __bf16* ap = ctx + (size_t)mRow * 512 + kk + aoff;
        a.h[0] = *(const v8bf*)(ap);
        a.h[1] = *(const v8bf*)(ap + 16);
#pragma unroll
        for (int nt = 0; nt < 4; ++nt) {
            const int n = n0 + nt * 16 + (lane & 15);
            AB16 b;
            const __bf16* bp = wT + (size_t)n * 512 + kk + boff;
            b.h[0] = *(const v8bf*)(bp);
            b.h[1] = *(const v8bf*)(bp + 16);
            acc[nt] = wmma_bf16(a.v, b.v, acc[nt]);
        }
    }
#pragma unroll
    for (int nt = 0; nt < 4; ++nt) {
        const int n  = n0 + nt * 16 + (lane & 15);
        const float bv = bias[n];
#pragma unroll
        for (int r = 0; r < 8; ++r) {
            const int mg = m0 + rbase + r;
            out[(size_t)mg * 512 + n] = acc[nt][r] + bv;
        }
    }
}

// ---------------------------------------------------------------------------
// Launcher
// ---------------------------------------------------------------------------
extern "C" void kernel_launch(void* const* d_in, const int* in_sizes, int n_in,
                              void* d_out, int out_size, void* d_ws, size_t ws_size,
                              hipStream_t stream) {
    const float* x       = (const float*)d_in[0];   // [16,1024,512]
    const float* W_qkv   = (const float*)d_in[1];   // [512,1536]
    const float* b_qkv   = (const float*)d_in[2];   // [1536]
    const float* W_proj  = (const float*)d_in[3];   // [512,512]
    const float* b_proj  = (const float*)d_in[4];   // [512]
    const float* lam     = (const float*)d_in[5];   // scalar
    const float* adj     = (const float*)d_in[6];   // [1024,1024]

    float* out_main = (float*)d_out;                              // [B,C,D]
    float* attn_out = (float*)d_out + (size_t)B_ * C_ * D_;       // [B,H,C,C]

    // workspace layout (bf16 buffers)
    char* ws = (char*)d_ws;
    __bf16* xbf    = (__bf16*)(ws);                               // 16 MB
    __bf16* wqkvT  = (__bf16*)(ws + 16777216);                    // 1.5 MB
    __bf16* wprojT = (__bf16*)(ws + 18350080);                    // 0.5 MB
    __bf16* qs     = (__bf16*)(ws + 18874368);                    // 16 MB
    __bf16* ks     = (__bf16*)(ws + 35651584);                    // 16 MB
    __bf16* vT     = (__bf16*)(ws + 52428800);                    // 16 MB
    __bf16* ctx    = (__bf16*)(ws + 69206016);                    // 16 MB

    // Stage 1: precision conversion + weight transposition
    {
        const int n = B_ * C_ * D_;
        k_convert_bf16<<<(n + 255) / 256, 256, 0, stream>>>(x, xbf, n);
        const int nq = 512 * 1536;
        k_transpose_bf16<<<(nq + 255) / 256, 256, 0, stream>>>(W_qkv, wqkvT, 512, 1536);
        const int np = 512 * 512;
        k_transpose_bf16<<<(np + 255) / 256, 256, 0, stream>>>(W_proj, wprojT, 512, 512);
    }
    // Stage 2: QKV projection (M=16384, N=1536, K=512)
    k_qkv_gemm<<<dim3(1536 / 64, 16384 / 64), 128, 0, stream>>>(
        xbf, wqkvT, b_qkv, qs, ks, vT);
    // Stage 3: fused biased attention + softmax + PV
    k_attn<<<dim3(C_ / 16, B_ * H_), 256, 0, stream>>>(
        qs, ks, vT, adj, lam, attn_out, ctx);
    // Stage 4: output projection (M=16384, N=512, K=512)
    k_proj_gemm<<<dim3(512 / 64, 16384 / 64), 128, 0, stream>>>(
        ctx, wprojT, b_proj, out_main);
}
